// MultiRigidSidechain_90967407329451
// MI455X (gfx1250) — matrix operand order repaired and checked by reading, not compiled, not run on hardware
//
#include <hip/hip_runtime.h>
#include <stdint.h>

// ---------------------------------------------------------------------------
// MultiRigidSidechain for MI455X (gfx1250)
// bf16 WMMA GEMMs (double-buffered LDS pipeline) + fp32 geometry epilogues
// ---------------------------------------------------------------------------

typedef __attribute__((ext_vector_type(16))) __bf16 v16bf;
typedef __attribute__((ext_vector_type(8)))  float  v8f;

static constexpr int kN   = 32768;
static constexpr int kDim = 1024;
static constexpr int BM = 128, BN = 128, BK = 32;
static constexpr int NSTEP = kDim / BK;   // 32

__device__ __forceinline__ unsigned short f2bf(float f) {
  union { float f; unsigned u; } v; v.f = f;
  unsigned r = v.u + 0x7FFFu + ((v.u >> 16) & 1u);   // round-to-nearest-even
  return (unsigned short)(r >> 16);
}

union FragBF { unsigned u[8]; v16bf v; };

// Stage one 16-ushort (32B) strip of the A tile into two uint4 registers.
// A_BF16: direct bf16 copy.  else: relu(A0)+relu(A1) in fp32 -> bf16 pack.
template<bool A_BF16>
__device__ __forceinline__ void stage_a(const float* p0, const float* p1,
                                        const unsigned short* pb, int kt,
                                        uint4& r0, uint4& r1)
{
  if (A_BF16) {
    const unsigned short* p = pb + kt;
    r0 = *(const uint4*)(p);
    r1 = *(const uint4*)(p + 8);
  } else {
    float vals[16];
    #pragma unroll
    for (int q = 0; q < 4; ++q) {
      float4 v = *(const float4*)(p0 + kt + 4*q);
      float4 w = *(const float4*)(p1 + kt + 4*q);
      vals[4*q+0] = fmaxf(v.x, 0.f) + fmaxf(w.x, 0.f);
      vals[4*q+1] = fmaxf(v.y, 0.f) + fmaxf(w.y, 0.f);
      vals[4*q+2] = fmaxf(v.z, 0.f) + fmaxf(w.z, 0.f);
      vals[4*q+3] = fmaxf(v.w, 0.f) + fmaxf(w.w, 0.f);
    }
    unsigned pk[8];
    #pragma unroll
    for (int i = 0; i < 8; ++i)
      pk[i] = (unsigned)f2bf(vals[2*i]) | ((unsigned)f2bf(vals[2*i+1]) << 16);
    r0 = make_uint4(pk[0], pk[1], pk[2], pk[3]);
    r1 = make_uint4(pk[4], pk[5], pk[6], pk[7]);
  }
}

// ---------------------------------------------------------------------------
// Tiled GEMM: C[32768,1024] = A @ W(bf16) + bias*scale (+ resid)
//  - block tile 128x128, K-step 32, double-buffered LDS (one barrier/step)
//  - 8 waves (4M x 2N); each wave: 2x4 v_wmma_f32_16x16x32_bf16 accumulators
//  - OUT_F32: fp32 result; OUT_BF16: bf16(relu(result)) for the next GEMM
// ---------------------------------------------------------------------------
template<bool A_BF16, bool HAS_RES, bool OUT_F32, bool OUT_BF16>
__global__ __launch_bounds__(256) void gemm_bf16_wmma(
    const float* __restrict__ A0f, const float* __restrict__ A1f,
    const unsigned short* __restrict__ Ab,
    const unsigned short* __restrict__ W,       // bf16 bits, row-major [K][N]
    const float* __restrict__ bias, float bias_scale,
    const float* __restrict__ resid,
    float* __restrict__ Cf, unsigned short* __restrict__ Cb)
{
  __shared__ __align__(16) unsigned short lA[2][BM][BK];   // 2 x 8 KB
  __shared__ __align__(16) unsigned short lB[2][BK][BN];   // 2 x 8 KB

  const int bm = blockIdx.y * BM;
  const int bn = blockIdx.x * BN;
  const int tid  = threadIdx.x;
  const int lane = tid & 31;
  const int wave = tid >> 5;
  const int wm = (wave & 3) * 32;   // wave M offset in block
  const int wn = (wave >> 2) * 64;  // wave N offset in block

  v8f acc[2][4] = {};

  // global->LDS staging assignments (each thread: one 32B strip of A and B)
  const int arow = tid >> 1;          // 0..127
  const int acol = (tid & 1) * 16;    // element units
  const int brow = tid >> 3;          // 0..31
  const int bcol = (tid & 7) * 16;

  // fragment lane mapping (ISA 7.12.2, 16-bit A 16x32 / 32-bit C 16x16)
  const int moff = lane & 15;
  const int koff = (lane >> 4) << 3;  // 0 or 8: A K-offset AND C M-offset

  // pre-offset source pointers for staging
  const float*          pA0 = A0f ? A0f + (size_t)(bm + arow) * kDim + acol : nullptr;
  const float*          pA1 = A1f ? A1f + (size_t)(bm + arow) * kDim + acol : nullptr;
  const unsigned short* pAb = Ab  ? Ab  + (size_t)(bm + arow) * kDim + acol : nullptr;
  const unsigned short* pB  = W + (size_t)brow * kDim + bn + bcol;

  // ---- prologue: stage tile 0 into buffer 0 ----
  uint4 a0, a1, b0, b1;
  stage_a<A_BF16>(pA0, pA1, pAb, 0, a0, a1);
  b0 = *(const uint4*)(pB);
  b1 = *(const uint4*)(pB + 8);
  *(uint4*)&lA[0][arow][acol]     = a0;
  *(uint4*)&lA[0][arow][acol + 8] = a1;
  *(uint4*)&lB[0][brow][bcol]     = b0;
  *(uint4*)&lB[0][brow][bcol + 8] = b1;
  __syncthreads();

  for (int s = 0; s < NSTEP; ++s) {
    const int cur = s & 1;
    const int nxt = cur ^ 1;

    // ---- issue global loads for step s+1 early (overlap with WMMA) ----
    if (s + 1 < NSTEP) {
      const int kt = (s + 1) * BK;
      stage_a<A_BF16>(pA0, pA1, pAb, kt, a0, a1);
      b0 = *(const uint4*)(pB + (size_t)kt * kDim);
      b1 = *(const uint4*)(pB + (size_t)kt * kDim + 8);
    }
    if (s + 2 < NSTEP) {     // L2 prefetch two steps ahead (global_prefetch_b8)
      const int kp = (s + 2) * BK;
      if (A_BF16) __builtin_prefetch(pAb + kp, 0, 1);
      else        { __builtin_prefetch(pA0 + kp, 0, 1); __builtin_prefetch(pA1 + kp, 0, 1); }
      __builtin_prefetch(pB + (size_t)kp * kDim, 0, 1);
    }

    // ---- fragments from LDS buffer `cur` (ds_load_b128 pairs) ----
    FragBF aF[2], bF[4];
    #pragma unroll
    for (int i = 0; i < 2; ++i) {
      const unsigned short* pr = &lA[cur][wm + i*16 + moff][0];
      uint4 lo = *(const uint4*)(pr + koff);        // K 0..7  (+lane-half 8)
      uint4 hi = *(const uint4*)(pr + 16 + koff);   // K 16..23(+lane-half 8)
      aF[i].u[0]=lo.x; aF[i].u[1]=lo.y; aF[i].u[2]=lo.z; aF[i].u[3]=lo.w;
      aF[i].u[4]=hi.x; aF[i].u[5]=hi.y; aF[i].u[6]=hi.z; aF[i].u[7]=hi.w;
    }
    #pragma unroll
    for (int j = 0; j < 4; ++j) {
      const int nc = wn + j*16 + moff;  // lane%16 == N column
      #pragma unroll
      for (int p = 0; p < 8; ++p) {
        const int k0 = (p < 4) ? (koff + 2*p) : (16 + koff + 2*(p - 4));
        bF[j].u[p] = (unsigned)lB[cur][k0][nc] | ((unsigned)lB[cur][k0 + 1][nc] << 16);
      }
    }

    #pragma unroll
    for (int i = 0; i < 2; ++i)
      #pragma unroll
      for (int j = 0; j < 4; ++j)
        acc[i][j] = __builtin_amdgcn_wmma_f32_16x16x32_bf16(
            false, aF[i].v, false, bF[j].v, (short)0, acc[i][j], false, false);

    // ---- commit staged registers into buffer `nxt` ----
    if (s + 1 < NSTEP) {
      *(uint4*)&lA[nxt][arow][acol]     = a0;
      *(uint4*)&lA[nxt][arow][acol + 8] = a1;
      *(uint4*)&lB[nxt][brow][bcol]     = b0;
      *(uint4*)&lB[nxt][brow][bcol + 8] = b1;
    }
    __syncthreads();
  }

  // ---- epilogue: bias (+residual); fp32 and/or bf16(relu) stores ----
  #pragma unroll
  for (int i = 0; i < 2; ++i) {
    #pragma unroll
    for (int j = 0; j < 4; ++j) {
      const int n = bn + wn + j*16 + moff;
      const float bv = bias[n] * bias_scale;
      #pragma unroll
      for (int r = 0; r < 8; ++r) {
        const int m = bm + wm + i*16 + r + koff;
        float v = acc[i][j][r] + bv;
        if (HAS_RES) v += resid[(size_t)m * kDim + n];
        if (OUT_F32)  Cf[(size_t)m * kDim + n] = v;
        if (OUT_BF16) Cb[(size_t)m * kDim + n] = f2bf(fmaxf(v, 0.f));
      }
    }
  }
}

// ---------------------------------------------------------------------------
__global__ __launch_bounds__(256) void cvt_bf16_kernel(
    const float* __restrict__ in, unsigned short* __restrict__ out, int n)
{
  const int i = blockIdx.x * blockDim.x + threadIdx.x;
  if (i < n) out[i] = f2bf(in[i]);
}

// ---------------------------------------------------------------------------
// ang = relu(act) @ w_ang + b_ang -> pairwise L2 normalize -> angles [N,7,2]
// ---------------------------------------------------------------------------
__global__ __launch_bounds__(256) void angles_kernel(
    const float* __restrict__ act, const float* __restrict__ w_ang,
    const float* __restrict__ b_ang, float* __restrict__ out_ang)
{
  const int n = blockIdx.x * blockDim.x + threadIdx.x;
  if (n >= kN) return;
  float acc[14];
  #pragma unroll
  for (int c = 0; c < 14; ++c) acc[c] = b_ang[c];
  const float* row = act + (size_t)n * kDim;
  for (int k = 0; k < kDim; k += 4) {
    const float4 a4 = *(const float4*)(row + k);
    const float a[4] = { fmaxf(a4.x, 0.f), fmaxf(a4.y, 0.f),
                         fmaxf(a4.z, 0.f), fmaxf(a4.w, 0.f) };
    #pragma unroll
    for (int q = 0; q < 4; ++q) {
      const float* w = w_ang + (k + q) * 14;   // uniform address -> scalar loads
      #pragma unroll
      for (int c = 0; c < 14; ++c) acc[c] = fmaf(a[q], w[c], acc[c]);
    }
  }
  float* o = out_ang + (size_t)n * 14;
  #pragma unroll
  for (int p = 0; p < 7; ++p) {
    const float s = acc[2*p], c = acc[2*p+1];
    const float inv = rsqrtf(fmaxf(s*s + c*c, 1e-12f));
    o[2*p]   = s * inv;
    o[2*p+1] = c * inv;
  }
}

// ---------------------------------------------------------------------------
// Rigid-frame chain + atom14 placement. Affine 3x4 (implicit row3 = 0,0,0,1).
// ---------------------------------------------------------------------------
__device__ __forceinline__ void compose_affine(const float* A, const float* B, float* O) {
  #pragma unroll
  for (int r = 0; r < 3; ++r) {
    const float a0 = A[r*4+0], a1 = A[r*4+1], a2 = A[r*4+2], a3 = A[r*4+3];
    #pragma unroll
    for (int c = 0; c < 3; ++c)
      O[r*4+c] = a0*B[0*4+c] + a1*B[1*4+c] + a2*B[2*4+c];
    O[r*4+3] = a0*B[3] + a1*B[7] + a2*B[11] + a3;
  }
}

__global__ __launch_bounds__(256) void frames_kernel(
    const float* __restrict__ angles, const float* __restrict__ rigids,
    const float* __restrict__ deff,   const float* __restrict__ lit,
    const float* __restrict__ mask,   const int* __restrict__ aatype,
    const int* __restrict__ gidx,
    float* __restrict__ out_pred, float* __restrict__ out_frames)
{
  const int n = blockIdx.x * blockDim.x + threadIdx.x;
  if (n >= kN) return;
  const int aa = aatype[n];

  float sn[8], cs[8];
  sn[0] = 0.f; cs[0] = 1.f;
  #pragma unroll
  for (int p = 0; p < 7; ++p) {
    sn[p+1] = angles[(size_t)n*14 + 2*p];
    cs[p+1] = angles[(size_t)n*14 + 2*p + 1];
  }

  // frames[i] = default_frame @ rotx(theta_i): only columns 1,2 mix.
  float F[8][12];
  #pragma unroll
  for (int i = 0; i < 8; ++i) {
    const float* D = deff + ((size_t)aa*8 + i) * 16;
    const float c = cs[i], s = sn[i];
    #pragma unroll
    for (int r = 0; r < 3; ++r) {
      F[i][r*4+0] = D[r*4+0];
      F[i][r*4+1] =  c*D[r*4+1] + s*D[r*4+2];
      F[i][r*4+2] = -s*D[r*4+1] + c*D[r*4+2];
      F[i][r*4+3] = D[r*4+3];
    }
  }

  // chi chain: f5 = f4@f5, f6 = f5@f6, f7 = f6@f7
  float T[12];
  compose_affine(F[4], F[5], T);
  #pragma unroll
  for (int q = 0; q < 12; ++q) F[5][q] = T[q];
  compose_affine(F[5], F[6], T);
  #pragma unroll
  for (int q = 0; q < 12; ++q) F[6][q] = T[q];
  compose_affine(F[6], F[7], T);
  #pragma unroll
  for (int q = 0; q < 12; ++q) F[7][q] = T[q];

  // global frames = backbone rigid @ frames_to_bb; write out (row3 = 0,0,0,1)
  const float* rg = rigids + (size_t)n * 16;
  #pragma unroll
  for (int i = 0; i < 8; ++i) {
    float G[12];
    compose_affine(rg, F[i], G);
    float* of = out_frames + ((size_t)n*8 + i) * 16;
    #pragma unroll
    for (int q = 0; q < 12; ++q) of[q] = G[q];
    of[12] = 0.f; of[13] = 0.f; of[14] = 0.f; of[15] = 1.f;
  }

  // atom14 placement: gather group frame back (same-wave store->load is ordered)
  #pragma unroll
  for (int a = 0; a < 14; ++a) {
    const int g = gidx[aa*14 + a];
    const float* Fg = out_frames + ((size_t)n*8 + g) * 16;
    const float px = lit[((size_t)aa*14 + a)*3 + 0];
    const float py = lit[((size_t)aa*14 + a)*3 + 1];
    const float pz = lit[((size_t)aa*14 + a)*3 + 2];
    const float mk = mask[aa*14 + a];
    float* op = out_pred + ((size_t)n*14 + a) * 3;
    op[0] = (Fg[0]*px + Fg[1]*py + Fg[2] *pz + Fg[3] ) * mk;
    op[1] = (Fg[4]*px + Fg[5]*py + Fg[6] *pz + Fg[7] ) * mk;
    op[2] = (Fg[8]*px + Fg[9]*py + Fg[10]*pz + Fg[11]) * mk;
  }
}

// ---------------------------------------------------------------------------
extern "C" void kernel_launch(void* const* d_in, const int* in_sizes, int n_in,
                              void* d_out, int out_size, void* d_ws, size_t ws_size,
                              hipStream_t stream)
{
  (void)in_sizes; (void)n_in; (void)out_size; (void)ws_size;

  const float* rep0   = (const float*)d_in[0];
  const float* rep1   = (const float*)d_in[1];
  const float* w_in   = (const float*)d_in[2];
  const float* b_in   = (const float*)d_in[3];
  const float* w1     = (const float*)d_in[4];
  const float* b1     = (const float*)d_in[5];
  const float* w2     = (const float*)d_in[6];
  const float* b2     = (const float*)d_in[7];
  const float* w_ang  = (const float*)d_in[8];
  const float* b_ang  = (const float*)d_in[9];
  const float* rigids = (const float*)d_in[10];
  const float* deff   = (const float*)d_in[11];
  const float* lit    = (const float*)d_in[12];
  const float* amask  = (const float*)d_in[13];
  const int*   aat    = (const int*)d_in[14];
  const int*   gidx   = (const int*)d_in[15];

  float* out_ang    = (float*)d_out;                    // [N,7,2]
  float* out_pred   = out_ang + (size_t)kN * 14;        // [N,14,3]
  float* out_frames = out_pred + (size_t)kN * 42;       // [N,8,4,4]

  const size_t actElems = (size_t)kN * kDim;
  float* ACT0 = (float*)d_ws;                        // fp32 act (residual src)
  float* ACT1 = ACT0 + actElems;                     // fp32 act (residual src)
  unsigned short* ACTb = (unsigned short*)(ACT1 + actElems); // bf16 relu(act)
  unsigned short* Hb   = ACTb + actElems;                    // bf16 relu(h)
  unsigned short* Wb0  = Hb + actElems;
  unsigned short* Wb1  = Wb0 + (size_t)kDim * kDim;
  unsigned short* Wb2  = Wb1 + (size_t)kDim * kDim;

  {
    const int n = kDim * kDim;
    const int blocks = (n + 255) / 256;
    cvt_bf16_kernel<<<blocks, 256, 0, stream>>>(w_in, Wb0, n);
    cvt_bf16_kernel<<<blocks, 256, 0, stream>>>(w1,   Wb1, n);
    cvt_bf16_kernel<<<blocks, 256, 0, stream>>>(w2,   Wb2, n);
  }

  dim3 grid(kDim / BN, kN / BM);   // (8, 256) blocks of 256 threads
  // act = (relu(rep0)+relu(rep1)) @ W_in + 2*b_in          -> ACT0 (fp32), ACTb (bf16 relu)
  gemm_bf16_wmma<false, false, true,  true ><<<grid, 256, 0, stream>>>(
      rep0, rep1, nullptr, Wb0, b_in, 2.0f, nullptr, ACT0, ACTb);
  // recycle 1: h = relu(act)@W1+b1 (bf16-only)             -> Hb
  gemm_bf16_wmma<true,  false, false, true ><<<grid, 256, 0, stream>>>(
      nullptr, nullptr, ACTb, Wb1, b1, 1.0f, nullptr, nullptr, Hb);
  //            act = act + relu(h)@W2+b2                   -> ACT1, ACTb
  gemm_bf16_wmma<true,  true,  true,  true ><<<grid, 256, 0, stream>>>(
      nullptr, nullptr, Hb, Wb2, b2, 1.0f, ACT0, ACT1, ACTb);
  // recycle 2
  gemm_bf16_wmma<true,  false, false, true ><<<grid, 256, 0, stream>>>(
      nullptr, nullptr, ACTb, Wb1, b1, 1.0f, nullptr, nullptr, Hb);
  gemm_bf16_wmma<true,  true,  true,  false><<<grid, 256, 0, stream>>>(
      nullptr, nullptr, Hb, Wb2, b2, 1.0f, ACT1, ACT0, nullptr);

  angles_kernel<<<kN / 256, 256, 0, stream>>>(ACT0, w_ang, b_ang, out_ang);
  frames_kernel<<<kN / 256, 256, 0, stream>>>(out_ang, rigids, deff, lit, amask,
                                              aat, gidx, out_pred, out_frames);
}
